// GaussianRasterizer_1142461301242
// MI455X (gfx1250) — matrix operand order repaired and checked by reading, compiled, not run on hardware
//
#include <hip/hip_runtime.h>

typedef __attribute__((ext_vector_type(16))) _Float16 v16h;
typedef __attribute__((ext_vector_type(8)))  float    v8f;
typedef __attribute__((ext_vector_type(4)))  unsigned int u32x4;
typedef __attribute__((ext_vector_type(8)))  int      i32x8;
typedef __attribute__((ext_vector_type(4)))  int      i32x4;

typedef __attribute__((address_space(1))) i32x4 gv4i;   // global vec4i
typedef __attribute__((address_space(3))) i32x4 sv4i;   // LDS vec4i
typedef __attribute__((address_space(3))) void  svoid;  // LDS void

#define NG    2048
#define HIMG  128
#define WIMG  128
#define NPIX  (HIMG*WIMG)
#define TANF  0.5f
#define ZNEARF 0.2f
#define AMINF (1.0f/255.0f)
#define AMAXF 0.99f

// workspace layout (float offsets)
#define OFF_TZ  0
#define OFF_PX  (NG)
#define OFF_PY  (2*NG)
#define OFF_CA  (3*NG)
#define OFF_CB  (4*NG)
#define OFF_CC  (5*NG)
#define OFF_OP  (6*NG)
#define OFF_CR  (7*NG)
#define OFF_CG  (8*NG)
#define OFF_CB2 (9*NG)
#define OFF_SA  (10*NG)   // float4[NG]  : px,py,conA,conB  (sorted)
#define OFF_SB  (14*NG)   // float2[NG]  : conC,opac        (sorted)
#define OFF_SC  (16*NG)   // _Float16[3*NG] : r,g,b         (sorted)

#define GEO_BYTES  (NG*16 + NG*8)       // 49152: sA + sB contiguous
#define COL_BYTES  (NG*3*2)             // 12288: sC
#define SMEM_BYTES (GEO_BYTES + COL_BYTES)

#if __has_builtin(__builtin_amdgcn_tensor_load_to_lds) && \
    __has_builtin(__builtin_amdgcn_s_wait_tensorcnt)
#define HAVE_TDM 1
#else
#define HAVE_TDM 0
#endif

#if __has_builtin(__builtin_amdgcn_global_load_async_to_lds_b128)
#define HAVE_ASYNC 1
#else
#define HAVE_ASYNC 0
#endif

// ---------------------------------------------------------------------------
// Kernel 1: per-gaussian preprocess (conic, screen pos, depth key)
// ---------------------------------------------------------------------------
__global__ __launch_bounds__(256) void gs_preprocess(
    const float* __restrict__ means3D, const float* __restrict__ opac,
    const float* __restrict__ cols,    const float* __restrict__ scales,
    const float* __restrict__ rots,    const float* __restrict__ vm,
    float* __restrict__ ws)
{
  int g = blockIdx.x * 256 + threadIdx.x;
  if (g >= NG) return;

  float qr = rots[4*g+0], qx = rots[4*g+1], qy = rots[4*g+2], qz = rots[4*g+3];
  float inv = rsqrtf(qr*qr + qx*qx + qy*qy + qz*qz);
  qr *= inv; qx *= inv; qy *= inv; qz *= inv;

  float R00 = 1.f-2.f*(qy*qy+qz*qz), R01 = 2.f*(qx*qy-qr*qz), R02 = 2.f*(qx*qz+qr*qy);
  float R10 = 2.f*(qx*qy+qr*qz), R11 = 1.f-2.f*(qx*qx+qz*qz), R12 = 2.f*(qy*qz-qr*qx);
  float R20 = 2.f*(qx*qz-qr*qy), R21 = 2.f*(qy*qz+qr*qx), R22 = 1.f-2.f*(qx*qx+qy*qy);

  float s0 = scales[3*g+0], s1 = scales[3*g+1], s2 = scales[3*g+2]; // SCALE_MOD = 1
  float M00=R00*s0, M01=R01*s1, M02=R02*s2;
  float M10=R10*s0, M11=R11*s1, M12=R12*s2;
  float M20=R20*s0, M21=R21*s1, M22=R22*s2;

  // cov3D = M * M^T (symmetric)
  float C00=M00*M00+M01*M01+M02*M02;
  float C01=M00*M10+M01*M11+M02*M12;
  float C02=M00*M20+M01*M21+M02*M22;
  float C11=M10*M10+M11*M11+M12*M12;
  float C12=M10*M20+M11*M21+M12*M22;
  float C22=M20*M20+M21*M21+M22*M22;

  float m0 = means3D[3*g+0], m1 = means3D[3*g+1], m2 = means3D[3*g+2];
  float t0 = vm[0]*m0 + vm[1]*m1 + vm[2]*m2  + vm[3];
  float t1 = vm[4]*m0 + vm[5]*m1 + vm[6]*m2  + vm[7];
  float t2 = vm[8]*m0 + vm[9]*m1 + vm[10]*m2 + vm[11];

  const float fx = WIMG/(2.f*TANF), fy = HIMG/(2.f*TANF);
  const float limx = 1.3f*TANF, limy = 1.3f*TANF;
  float invz = 1.f/t2;
  float txz = fminf(fmaxf(t0*invz, -limx), limx) * t2;
  float tyz = fminf(fmaxf(t1*invz, -limy), limy) * t2;

  float J00 = fx*invz, J02 = -fx*txz*invz*invz;
  float J11 = fy*invz, J12 = -fy*tyz*invz*invz;

  // T = J @ Wm (J01 = J10 = 0)
  float T00 = J00*vm[0] + J02*vm[8];
  float T01 = J00*vm[1] + J02*vm[9];
  float T02 = J00*vm[2] + J02*vm[10];
  float T10 = J11*vm[4] + J12*vm[8];
  float T11 = J11*vm[5] + J12*vm[9];
  float T12 = J11*vm[6] + J12*vm[10];

  // cov2D = T * C * T^T
  float U0 = T00*C00 + T01*C01 + T02*C02;
  float U1 = T00*C01 + T01*C11 + T02*C12;
  float U2 = T00*C02 + T01*C12 + T02*C22;
  float V0 = T10*C00 + T11*C01 + T12*C02;
  float V1 = T10*C01 + T11*C11 + T12*C12;
  float V2 = T10*C02 + T11*C12 + T12*C22;

  float a = U0*T00 + U1*T01 + U2*T02 + 0.3f;
  float b = U0*T10 + U1*T11 + U2*T12;
  float c = V0*T10 + V1*T11 + V2*T12 + 0.3f;

  float det   = a*c - b*b;
  float dsafe = (det > 0.f) ? det : 1.f;
  float conA  = c / dsafe, conB = -b / dsafe, conC = a / dsafe;
  float px = fx * t0 * invz + (WIMG - 1) * 0.5f;
  float py = fy * t1 * invz + (HIMG - 1) * 0.5f;
  bool valid = (det > 0.f) && (t2 > ZNEARF);

  ws[OFF_TZ  + g] = t2;
  ws[OFF_PX  + g] = px;
  ws[OFF_PY  + g] = py;
  ws[OFF_CA  + g] = conA;
  ws[OFF_CB  + g] = conB;
  ws[OFF_CC  + g] = conC;
  ws[OFF_OP  + g] = valid ? opac[g] : 0.f;   // folds ~valid into alpha gating
  ws[OFF_CR  + g] = cols[3*g+0];
  ws[OFF_CG  + g] = cols[3*g+1];
  ws[OFF_CB2 + g] = cols[3*g+2];
}

// ---------------------------------------------------------------------------
// Kernel 2: bitonic depth sort (2048 keys, one 1024-thread block) + gather
// ---------------------------------------------------------------------------
__global__ __launch_bounds__(1024) void gs_sort_gather(float* __restrict__ ws)
{
  __shared__ float sk[NG];
  __shared__ int   sv[NG];
  int tid = threadIdx.x;

  for (int i = tid; i < NG; i += 1024) { sk[i] = ws[OFF_TZ + i]; sv[i] = i; }
  __syncthreads();

  for (int size = 2; size <= NG; size <<= 1) {
    for (int stride = size >> 1; stride > 0; stride >>= 1) {
      for (int i = tid; i < NG; i += 1024) {
        int j = i ^ stride;
        if (j > i) {
          float ki = sk[i], kj = sk[j];
          bool up = ((i & size) == 0);
          bool sw = up ? (ki > kj) : (ki < kj);
          if (sw) {
            sk[i] = kj; sk[j] = ki;
            int t = sv[i]; sv[i] = sv[j]; sv[j] = t;
          }
        }
      }
      __syncthreads();
    }
  }

  float4*    wsA = (float4*)(ws + OFF_SA);
  float2*    wsB = (float2*)(ws + OFF_SB);
  _Float16*  wsC = (_Float16*)(ws + OFF_SC);
  for (int i = tid; i < NG; i += 1024) {
    int g = sv[i];
    wsA[i] = make_float4(ws[OFF_PX + g], ws[OFF_PY + g], ws[OFF_CA + g], ws[OFF_CB + g]);
    wsB[i] = make_float2(ws[OFF_CC + g], ws[OFF_OP + g]);
    wsC[3*i + 0] = (_Float16)ws[OFF_CR  + g];
    wsC[3*i + 1] = (_Float16)ws[OFF_CG  + g];
    wsC[3*i + 2] = (_Float16)ws[OFF_CB2 + g];
  }
}

// ---------------------------------------------------------------------------
// Kernel 3: rasterize. 256 threads = 8 waves, each wave owns 16 pixels.
// Staging: TDM tensor_load_to_lds for the 48 KB geometry block (1-D tile),
// async global->LDS b128 for the 12 KB f16 color table. Blend reduction via
// v_wmma_f32_16x16x32_f16.
// ---------------------------------------------------------------------------
__global__ __launch_bounds__(256) void gs_raster(
    const float4* __restrict__ gA, const float2* __restrict__ gB,
    const _Float16* __restrict__ gC, const float* __restrict__ bg,
    float* __restrict__ out)
{
  __shared__ __align__(16) unsigned char smem[SMEM_BYTES];
  float4*   sA = (float4*)smem;                         // 32768 B
  float2*   sB = (float2*)(smem + NG*16);               // 16384 B
  _Float16* sC = (_Float16*)(smem + GEO_BYTES);         // 12288 B

  int tid  = threadIdx.x;
  int lane = tid & 31;
  int wv   = tid >> 5;

#if HAVE_TDM
  if (wv == 0) {
    // D# for a 1-D tensor: 12288 dwords, global = sorted geometry block.
    unsigned long long ga = (unsigned long long)(const void*)gA;
    unsigned int lds_off  = (unsigned int)(unsigned long long)(svoid*)smem;
    const unsigned int ndw = GEO_BYTES / 4;  // 12288, fits 16-bit tile_dim0
    u32x4 g0;
    g0[0] = 1u;                                        // count = 1 (valid D#)
    g0[1] = lds_off;                                   // lds_addr (bytes)
    g0[2] = (unsigned int)(ga & 0xFFFFFFFFu);          // global_addr[31:0]
    g0[3] = (unsigned int)((ga >> 32) & 0x01FFFFFFu)   // global_addr[56:32]
            | (2u << 30);                              // type = 2 ("image")
    i32x8 g1;
    g1[0] = (int)(2u << 16);                           // data_size = 4 B
    g1[1] = (int)((ndw & 0xFFFFu) << 16);              // tensor_dim0[15:0]
    g1[2] = (int)(1u << 16);                           // tensor_dim0 hi=0, tensor_dim1 = 1
    g1[3] = (int)((ndw & 0xFFFFu) << 16);              // tile_dim0 = ndw
    g1[4] = 0;                                         // tile_dim1/2 unused
    g1[5] = (int)ndw;                                  // tensor_dim0_stride lo32
    g1[6] = (int)((ndw & 0xFFFFu) << 16);              // stride0 hi=0, stride1 lo16
    g1[7] = 0;
    i32x4 gz4 = {0, 0, 0, 0};
    i32x8 gz8 = {0, 0, 0, 0, 0, 0, 0, 0};
    __builtin_amdgcn_tensor_load_to_lds(g0, g1, gz4, gz4, gz8, 0);
    __builtin_amdgcn_s_wait_tensorcnt(0);
  }
#else
  for (int i = tid; i < NG; i += 256) { sA[i] = gA[i]; sB[i] = gB[i]; }
#endif

#if HAVE_ASYNC
  // 768 x b128 async transfers, 3 per thread (ASYNCcnt-tracked)
  for (int i = tid; i < COL_BYTES / 16; i += 256) {
    __builtin_amdgcn_global_load_async_to_lds_b128(
        (gv4i*)(gC + 8 * i), (sv4i*)(sC + 8 * i), 0, 0);
  }
#if __has_builtin(__builtin_amdgcn_s_wait_asynccnt)
  __builtin_amdgcn_s_wait_asynccnt(0);
#else
  asm volatile("s_wait_asynccnt 0" ::: "memory");
#endif
#else
  for (int i = tid; i < 3*NG; i += 256) { sC[i] = gC[i]; }
#endif
  __syncthreads();

  int h    = lane >> 4;        // lane half (A/B layout half)
  int p    = lane & 15;        // pixel-in-tile for the scan; channel for store
  int base = blockIdx.x * 128 + wv * 16;
  int pid  = base + p;
  float xf = (float)(pid & (WIMG - 1));
  float yf = (float)(pid >> 7);

  float T = 1.0f;              // transmittance for pixel p
  v8f acc = {};                // WMMA C/D accumulator (16x16 f32)

  for (int ch = 0; ch < NG / 32; ++ch) {
    int bk = ch << 5;
    v16h a = {};
#pragma unroll
    for (int k = 0; k < 32; ++k) {
      float4 A4 = sA[bk + k];  // px, py, conA, conB  (LDS broadcast)
      float2 B2 = sB[bk + k];  // conC, opac
      float dx = xf - A4.x;
      float dy = yf - A4.y;
      float pw = -0.5f * (A4.z * dx * dx + B2.x * dy * dy) - A4.w * dx * dy;
      float al = B2.y * __expf(pw);
      al = fminf(al, AMAXF);
      al = (pw > 0.f || al < AMINF) ? 0.f : al;
      float w = T * al;
      T = T - w;               // T *= (1 - alpha)
      // A layout: element j <-> K = (j&7) + 16*(j>=8) + 8*h
      const int j = (k & 7) | ((k >> 4) << 3);
      bool mine = (((k >> 3) & 1) == h);
      a[j] = mine ? (_Float16)w : a[j];
    }
    // B layout: element j of lane (h, n=lane&15) holds cols[K=(j&7)+16*(j>=8)+8h][n]
    v16h b = {};
#pragma unroll
    for (int j = 0; j < 16; ++j) {
      int kk = bk + (j & 7) + ((j >> 3) << 4) + (h << 3);
      _Float16 cv = sC[3 * kk + (p < 3 ? p : 0)];
      b[j] = (p < 3) ? cv : (_Float16)0.0f;
    }
    acc = __builtin_amdgcn_wmma_f32_16x16x32_f16(
        /*neg_a=*/false, a, /*neg_b=*/false, b,
        /*c_mod=*/(short)0, acc, /*reuse_a=*/false, /*reuse_b=*/false);
  }

  // Fetch final transmittance for rows m = 8h..8h+7 (all lanes participate)
  float tvals[8];
#pragma unroll
  for (int m0 = 0; m0 < 8; ++m0)
    tvals[m0] = __shfl(T, (h << 3) + m0, 32);

  // D layout: element m0 of lane (h, n) = D[m = m0 + 8h][channel n]
  if (p < 3) {
#pragma unroll
    for (int m0 = 0; m0 < 8; ++m0) {
      int m = (h << 3) + m0;
      out[p * NPIX + base + m] = acc[m0] + tvals[m0] * bg[p];
    }
  }
}

// ---------------------------------------------------------------------------
extern "C" void kernel_launch(void* const* d_in, const int* in_sizes, int n_in,
                              void* d_out, int out_size, void* d_ws, size_t ws_size,
                              hipStream_t stream)
{
  (void)in_sizes; (void)n_in; (void)out_size; (void)ws_size;
  const float* means3D = (const float*)d_in[0];
  // d_in[1] = means2D (unused by reference math)
  const float* opac    = (const float*)d_in[2];
  const float* cols    = (const float*)d_in[3];
  const float* scales  = (const float*)d_in[4];
  const float* rots    = (const float*)d_in[5];
  const float* bg      = (const float*)d_in[6];
  const float* vm      = (const float*)d_in[7];
  float* ws  = (float*)d_ws;
  float* out = (float*)d_out;

  gs_preprocess<<<NG / 256, 256, 0, stream>>>(means3D, opac, cols, scales, rots, vm, ws);
  gs_sort_gather<<<1, 1024, 0, stream>>>(ws);
  gs_raster<<<NPIX / 128, 256, 0, stream>>>(
      (const float4*)(ws + OFF_SA), (const float2*)(ws + OFF_SB),
      (const _Float16*)(ws + OFF_SC), bg, out);
}
